// TransformerCritic_8486855377305
// MI455X (gfx1250) — matrix operand order repaired and checked by reading, not compiled, not run on hardware
//
#include <hip/hip_runtime.h>
#include <hip/hip_bf16.h>

// ---------------------------------------------------------------------------
// TransformerCritic fused forward for MI455X (gfx1250), wave32 + WMMA f16.
// B=1024, N=32, OBS=128, ACT=16, H=128, FIN=256.
// One block per batch; all intermediates in LDS; all GEMMs on
// v_wmma_f32_16x16x32_f16; input tile staged via TDM (tensor_load_to_lds);
// weights converted to f16 once per launch into d_ws (L2-resident, shared by
// all 1024 blocks). Each weight B-fragment feeds 2 WMMAs (6 in the GRU).
// ---------------------------------------------------------------------------

typedef __attribute__((ext_vector_type(16))) _Float16     v16h;
typedef __attribute__((ext_vector_type(8)))  _Float16     v8h;
typedef __attribute__((ext_vector_type(8)))  float        v8f;
typedef __attribute__((ext_vector_type(4)))  unsigned int u32x4;
typedef __attribute__((ext_vector_type(8)))  int          i32x8;
typedef __attribute__((ext_vector_type(4)))  int          i32x4;

#define NWAVES 8

// f16 weight workspace offsets (in halves; all multiples of 16 -> 32B aligned)
#define OFF_WS   0                         // W_s   128x128
#define OFF_WSA  16384                     // W_sa  128x160 (padded from 144)
#define OFF_WK   (OFF_WSA + 20480)         // W_k   128x128
#define OFF_WQ   (OFF_WK  + 16384)         // W_q   128x128
#define OFF_WV   (OFF_WQ  + 16384)         // W_v   128x128
#define OFF_WE   (OFF_WV  + 16384)         // W_e   128x128
#define OFF_WI   (OFF_WE  + 16384)         // gru_Wi 768x256
#define OFF_W1   (OFF_WI  + 196608)        // W1    64x256
#define WS_HALVES (OFF_W1 + 16384)

__device__ __forceinline__ float lrelu(float x)    { return x > 0.f ? x : 0.01f * x; }
__device__ __forceinline__ float sigmoidf(float x) { return 1.f / (1.f + __expf(-x)); }

// A-fragment (16x32 f16): lanes 0-15 hold M=lane, K = {0..7, 16..23};
// lanes 16-31 hold M=lane-16, K = {8..15, 24..31}.
__device__ __forceinline__ v16h load_a_frag(const _Float16* tile, int ld) {
    const int lane = threadIdx.x & 31;
    const int r    = lane & 15;
    const int hsel = (lane >> 4) * 8;
    const _Float16* p = tile + r * ld;
    v8h lo = *(const v8h*)(p + hsel);
    v8h hi = *(const v8h*)(p + 16 + hsel);
    v16h out;
#pragma unroll
    for (int i = 0; i < 8; ++i) { out[i] = lo[i]; out[i + 8] = hi[i]; }
    return out;
}

// B-fragment (32x16 f16) from W stored row-major [N][K] (i.e. Bmat = W^T):
// lane n%16 supplies column n, K = {0..7, 16..23} (lanes 0-15) or {8..15, 24..31} (lanes 16-31).
__device__ __forceinline__ v16h load_b_frag(const _Float16* w, int ld) {
    const int lane = threadIdx.x & 31;
    const int n    = lane & 15;
    const int kof  = (lane >> 4) * 8;
    const _Float16* p = w + n * ld;
    v8h lo = *(const v8h*)(p + kof);
    v8h hi = *(const v8h*)(p + 16 + kof);
    v16h out;
#pragma unroll
    for (int i = 0; i < 8; ++i) { out[i] = lo[i]; out[i + 8] = hi[i]; }
    return out;
}

// C = act(A(32xK) @ W^T + bias): each wave owns an N-tile and computes both
// M-tiles so every B-fragment (weight data) feeds two WMMAs.
__device__ __forceinline__ void gemm32(const _Float16* __restrict__ A, int lda, int K,
                                       const _Float16* __restrict__ W, int ldw,
                                       const float* __restrict__ bias,
                                       _Float16* __restrict__ out, int ldo,
                                       int Ncols, int act, bool trans) {
    const int wave  = threadIdx.x >> 5;
    const int lane  = threadIdx.x & 31;
    const int n     = lane & 15;
    const int rbase = (lane >> 4) * 8;
    for (int nt = wave; nt < (Ncols >> 4); nt += NWAVES) {
        v8f acc0 = {}, acc1 = {};
        for (int kc = 0; kc < K; kc += 32) {
            v16h bm = load_b_frag(W + nt * 16 * ldw + kc, ldw);
            v16h a0 = load_a_frag(A + kc, lda);
            v16h a1 = load_a_frag(A + 16 * lda + kc, lda);
            acc0 = __builtin_amdgcn_wmma_f32_16x16x32_f16(false, a0, false, bm, (short)0, acc0, false, false);
            acc1 = __builtin_amdgcn_wmma_f32_16x16x32_f16(false, a1, false, bm, (short)0, acc1, false, false);
        }
        const float bv  = bias ? bias[nt * 16 + n] : 0.f;
        const int   col = nt * 16 + n;
#pragma unroll
        for (int v = 0; v < 8; ++v) {
            float x0 = acc0[v] + bv, x1 = acc1[v] + bv;
            if (act) { x0 = lrelu(x0); x1 = lrelu(x1); }
            const int r0 = rbase + v, r1 = 16 + rbase + v;
            if (trans) { out[col * ldo + r0] = (_Float16)x0; out[col * ldo + r1] = (_Float16)x1; }
            else       { out[r0 * ldo + col] = (_Float16)x0; out[r1 * ldo + col] = (_Float16)x1; }
        }
    }
}

// --------------------------- TDM input staging ------------------------------
#if __has_builtin(__builtin_amdgcn_tensor_load_to_lds)
#define HAVE_TDM 1
// 2D f32 tile load Global->LDS via the Tensor Data Mover (D# per ISA 8.3/8.4).
__device__ __forceinline__ void tdm_load_2d_f32(unsigned lds_off, const void* gptr,
                                                unsigned dim0, unsigned dim1,
                                                unsigned stride0) {
    const unsigned long long ga = (unsigned long long)(size_t)gptr;
    u32x4 g0;
    g0.x = 1u;                                               // count=1, user desc
    g0.y = lds_off;                                          // lds_addr (bytes)
    g0.z = (unsigned)ga;                                     // global_addr[31:0]
    g0.w = (unsigned)((ga >> 32) & 0x1FFFFFFu) | (2u << 30); // addr[56:32] | type=2
    i32x8 g1;
    g1[0] = 0x20000;                                  // data_size=4B, mask=0
    g1[1] = (int)(dim0 << 16);                        // tensor_dim0 (lo16)
    g1[2] = (int)((dim0 >> 16) | (dim1 << 16));       // dim0 hi | tensor_dim1 lo
    g1[3] = (int)((dim1 >> 16) | (dim0 << 16));       // dim1 hi | tile_dim0=dim0
    g1[4] = (int)dim1;                                // tile_dim1=dim1, tile_dim2=0
    g1[5] = (int)stride0;                             // tensor_dim0_stride lo32
    g1[6] = 0;
    g1[7] = 0;
    const i32x4 z4 = {0, 0, 0, 0};
#if __clang_major__ >= 23
    const i32x8 z8 = {0, 0, 0, 0, 0, 0, 0, 0};
    __builtin_amdgcn_tensor_load_to_lds(g0, g1, z4, z4, z8, 0);
#else
    __builtin_amdgcn_tensor_load_to_lds(g0, g1, z4, z4, 0);
#endif
}
#endif

// ------------------------- weight conversion kernels ------------------------
__global__ void cvt_f16_kernel(const float* __restrict__ in, _Float16* __restrict__ out, int n) {
    int i = blockIdx.x * blockDim.x + threadIdx.x;
    if (i < n) out[i] = (_Float16)in[i];
}

__global__ void cvt_wsa_pad_kernel(const float* __restrict__ in, _Float16* __restrict__ out) {
    int i = blockIdx.x * blockDim.x + threadIdx.x;   // 128 x 160
    if (i < 128 * 160) {
        int r = i / 160, k = i - r * 160;
        out[i] = (_Float16)(k < 144 ? in[r * 144 + k] : 0.f);
    }
}

// ------------------------------- main kernel --------------------------------
__global__ __launch_bounds__(256) void critic_fused_kernel(
    const float* __restrict__ states,   // [1024][32][128]
    const float* __restrict__ actions,  // [1024][32][16]
    const _Float16* __restrict__ w16,   // f16 weight workspace
    const float* __restrict__ b_sa, const float* __restrict__ b_s,
    const float* __restrict__ b_k,  const float* __restrict__ b_q,
    const float* __restrict__ b_v,  const float* __restrict__ b_e,
    const float* __restrict__ gru_bi, const float* __restrict__ gru_bh,
    const float* __restrict__ b1, const float* __restrict__ W2,
    const float* __restrict__ b2,
    float* __restrict__ out)            // [value | weight | h] concatenated
{
    const int b    = blockIdx.x;
    const int tid  = threadIdx.x;
    const int wave = tid >> 5;
    const int lane = tid & 31;

    float* out_value  = out;                       // 1024*32
    float* out_weight = out + 32768;               // 1024*32*31
    float* out_h      = out + 1048576;             // 1024*32*256

    const _Float16* Ws  = w16 + OFF_WS;
    const _Float16* Wsa = w16 + OFF_WSA;
    const _Float16* Wk  = w16 + OFF_WK;
    const _Float16* Wq  = w16 + OFF_WQ;
    const _Float16* Wv  = w16 + OFF_WV;
    const _Float16* We  = w16 + OFF_WE;
    const _Float16* Wi  = w16 + OFF_WI;
    const _Float16* W1w = w16 + OFF_W1;

    __shared__ __align__(32) float    s_stf32[32 * 128];  // TDM-staged states
    __shared__ __align__(32) float    s_actf32[32 * 16];  // TDM-staged actions
    __shared__ __align__(32) _Float16 sA[32 * 160];       // [states|actions|pad] f16
    __shared__ __align__(32) _Float16 s_es[32 * 128];
    __shared__ __align__(32) _Float16 s_esa[32 * 128];
    __shared__ __align__(32) _Float16 s_q[32 * 128];
    __shared__ __align__(32) _Float16 s_k[32 * 128];
    __shared__ __align__(32) _Float16 s_vT[128 * 32];     // vals transposed
    __shared__ __align__(32) float    s_score[32 * 32];
    __shared__ __align__(32) _Float16 s_wh[32 * 32];      // attention matrix (diag=0)
    __shared__ __align__(32) _Float16 s_node[32 * 256];   // [q_embed | x]
    __shared__ __align__(32) _Float16 s_h[32 * 256];
    __shared__ __align__(32) _Float16 s_fc1[32 * 64];

    // Warm the GRU weights (largest shared block) into cache.
    if (tid < 192) __builtin_prefetch(Wi + tid * 1024, 0, 1);

    // Stage 0: stage raw inputs into LDS (TDM DMA when available), then build
    // sa = [states|actions|0-pad] as f16.
#ifdef HAVE_TDM
    if (wave == 0) {   // TDM ignores EXEC -> issue from a single wave
        tdm_load_2d_f32((unsigned)(size_t)&s_stf32[0],  states  + (size_t)b * 4096, 128, 32, 128);
        tdm_load_2d_f32((unsigned)(size_t)&s_actf32[0], actions + (size_t)b * 512,   16, 32,  16);
        __builtin_amdgcn_s_wait_tensorcnt(0);
    }
    __syncthreads();
    for (int i = tid; i < 32 * 160; i += 256) {
        const int r = i / 160, k = i - r * 160;
        const float v = (k < 128) ? s_stf32[r * 128 + k]
                      : (k < 144) ? s_actf32[r * 16 + (k - 128)]
                      : 0.f;
        sA[i] = (_Float16)v;
    }
#else
    for (int i = tid; i < 32 * 160; i += 256) {
        const int r = i / 160, k = i - r * 160;
        float v;
        if (k < 128)      v = states[((size_t)b * 32 + r) * 128 + k];
        else if (k < 144) v = actions[((size_t)b * 32 + r) * 16 + (k - 128)];
        else              v = 0.f;
        sA[i] = (_Float16)v;
    }
#endif
    __syncthreads();

    // Stage 1: three projections off the raw inputs.
    gemm32(sA, 160, 128, Ws,  128, b_s,  s_es,   128, 128, 1, false); // e_s
    gemm32(sA, 160, 160, Wsa, 160, b_sa, s_esa,  128, 128, 1, false); // e_sa
    gemm32(sA, 160, 128, We,  128, b_e,  s_node, 256, 128, 1, false); // q_embed
    __syncthreads();

    // Stage 2: k/q/v projections (vals stored transposed for attention apply).
    gemm32(s_esa, 128, 128, Wk, 128, b_k, s_k,  128, 128, 0, false);
    gemm32(s_es,  128, 128, Wq, 128, b_q, s_q,  128, 128, 0, false);
    gemm32(s_esa, 128, 128, Wv, 128, b_v, s_vT, 32,  128, 0, true);
    __syncthreads();

    // Stage 3: score = q @ k^T / sqrt(H)  (32x32, K=128); 2 N-tiles, B reused.
    if (wave < 2) {
        const int nt = wave;
        const int n = lane & 15, rbase = (lane >> 4) * 8;
        v8f acc0 = {}, acc1 = {};
        for (int kc = 0; kc < 128; kc += 32) {
            v16h bm = load_b_frag(s_k + nt * 16 * 128 + kc, 128);
            v16h a0 = load_a_frag(s_q + kc, 128);
            v16h a1 = load_a_frag(s_q + 16 * 128 + kc, 128);
            acc0 = __builtin_amdgcn_wmma_f32_16x16x32_f16(false, a0, false, bm, (short)0, acc0, false, false);
            acc1 = __builtin_amdgcn_wmma_f32_16x16x32_f16(false, a1, false, bm, (short)0, acc1, false, false);
        }
#pragma unroll
        for (int v = 0; v < 8; ++v) {
            s_score[(rbase + v) * 32 + nt * 16 + n]      = acc0[v] * 0.08838834764831845f;
            s_score[(16 + rbase + v) * 32 + nt * 16 + n] = acc1[v] * 0.08838834764831845f;
        }
    }
    __syncthreads();

    // Stage 4: softmax over off-diagonal; emit weight + f16 attention matrix.
    for (int r = wave; r < 32; r += NWAVES) {
        const bool diag = (lane == r);
        const float sc  = s_score[r * 32 + lane];
        float val = diag ? -1e30f : sc;
        float m = val;
#pragma unroll
        for (int o = 16; o > 0; o >>= 1) m = fmaxf(m, __shfl_xor(m, o, 32));
        const float e = diag ? 0.f : __expf(val - m);
        float sum = e;
#pragma unroll
        for (int o = 16; o > 0; o >>= 1) sum += __shfl_xor(sum, o, 32);
        const float wgt = e / sum;
        s_wh[r * 32 + lane] = (_Float16)wgt;
        if (!diag) {
            const int jp = lane - (lane > r ? 1 : 0);
            out_weight[((size_t)b * 32 + r) * 31 + jp] = wgt;
        }
    }
    __syncthreads();

    // Stage 5: x = attn(32x32) @ vals(32x128) -> node[:,128:256].
    gemm32(s_wh, 32, 32, s_vT, 32, nullptr, s_node + 128, 256, 128, 0, false);
    __syncthreads();

    // Stage 6: GRU with h0 = 0 (gh collapses to gru_bh).
    // Per N-tile: 3 gate B-fragments x 2 A-fragments -> 6 WMMAs per B load set.
    {
        const int n = lane & 15, rbase = (lane >> 4) * 8;
        for (int nt = wave; nt < 16; nt += NWAVES) {
            v8f ar0 = {}, az0 = {}, an0 = {}, ar1 = {}, az1 = {}, an1 = {};
            for (int kc = 0; kc < 256; kc += 32) {
                v16h br = load_b_frag(Wi + (      nt * 16) * 256 + kc, 256);
                v16h bz = load_b_frag(Wi + (256 + nt * 16) * 256 + kc, 256);
                v16h bn = load_b_frag(Wi + (512 + nt * 16) * 256 + kc, 256);
                v16h a0 = load_a_frag(s_node + kc, 256);
                v16h a1 = load_a_frag(s_node + 16 * 256 + kc, 256);
                ar0 = __builtin_amdgcn_wmma_f32_16x16x32_f16(false, a0, false, br, (short)0, ar0, false, false);
                az0 = __builtin_amdgcn_wmma_f32_16x16x32_f16(false, a0, false, bz, (short)0, az0, false, false);
                an0 = __builtin_amdgcn_wmma_f32_16x16x32_f16(false, a0, false, bn, (short)0, an0, false, false);
                ar1 = __builtin_amdgcn_wmma_f32_16x16x32_f16(false, a1, false, br, (short)0, ar1, false, false);
                az1 = __builtin_amdgcn_wmma_f32_16x16x32_f16(false, a1, false, bz, (short)0, az1, false, false);
                an1 = __builtin_amdgcn_wmma_f32_16x16x32_f16(false, a1, false, bn, (short)0, an1, false, false);
            }
            const int c = nt * 16 + n;
            const float bir = gru_bi[c], biz = gru_bi[256 + c], bin = gru_bi[512 + c];
            const float bhr = gru_bh[c], bhz = gru_bh[256 + c], bhn = gru_bh[512 + c];
#pragma unroll
            for (int v = 0; v < 8; ++v) {
                {
                    const int row = rbase + v;
                    const float r  = sigmoidf(ar0[v] + bir + bhr);
                    const float z  = sigmoidf(az0[v] + biz + bhz);
                    const float nn = tanhf(an0[v] + bin + r * bhn);
                    const float h  = (1.f - z) * nn;            // + z*h0 (=0)
                    out_h[((size_t)b * 32 + row) * 256 + c] = h;
                    s_h[row * 256 + c] = (_Float16)h;
                }
                {
                    const int row = 16 + rbase + v;
                    const float r  = sigmoidf(ar1[v] + bir + bhr);
                    const float z  = sigmoidf(az1[v] + biz + bhz);
                    const float nn = tanhf(an1[v] + bin + r * bhn);
                    const float h  = (1.f - z) * nn;
                    out_h[((size_t)b * 32 + row) * 256 + c] = h;
                    s_h[row * 256 + c] = (_Float16)h;
                }
            }
        }
    }
    __syncthreads();

    // Stage 7: fc1 = lrelu(h @ W1^T + b1)   (32x64, K=256)
    gemm32(s_h, 256, 256, W1w, 256, b1, s_fc1, 64, 64, 1, false);
    __syncthreads();

    // Stage 8: value = fc1 @ W2^T + b2   (tiny 64-dot per row)
    {
        const int row = tid >> 3, part = tid & 7;
        float sum = 0.f;
#pragma unroll
        for (int c = part * 8; c < part * 8 + 8; ++c)
            sum += (float)s_fc1[row * 64 + c] * W2[c];
        sum += __shfl_down(sum, 4, 8);
        sum += __shfl_down(sum, 2, 8);
        sum += __shfl_down(sum, 1, 8);
        if (part == 0) out_value[(size_t)b * 32 + row] = sum + b2[0];
    }
}

// ------------------------------- launcher -----------------------------------
extern "C" void kernel_launch(void* const* d_in, const int* in_sizes, int n_in,
                              void* d_out, int out_size, void* d_ws, size_t ws_size,
                              hipStream_t stream) {
    const float* states  = (const float*)d_in[0];
    const float* actions = (const float*)d_in[1];
    const float* W_sa    = (const float*)d_in[2];
    const float* b_sa    = (const float*)d_in[3];
    const float* W_s     = (const float*)d_in[4];
    const float* b_s     = (const float*)d_in[5];
    const float* W_k     = (const float*)d_in[6];
    const float* b_k     = (const float*)d_in[7];
    const float* W_q     = (const float*)d_in[8];
    const float* b_q     = (const float*)d_in[9];
    const float* W_v     = (const float*)d_in[10];
    const float* b_v     = (const float*)d_in[11];
    const float* W_e     = (const float*)d_in[12];
    const float* b_e     = (const float*)d_in[13];
    const float* gru_Wi  = (const float*)d_in[14];
    // d_in[15] (gru_Wh) unused: h0 == 0 so gh == gru_bh.
    const float* gru_bi  = (const float*)d_in[16];
    const float* gru_bh  = (const float*)d_in[17];
    const float* W1      = (const float*)d_in[18];
    const float* b1      = (const float*)d_in[19];
    const float* W2      = (const float*)d_in[20];
    const float* b2      = (const float*)d_in[21];

    _Float16* w16 = (_Float16*)d_ws;   // needs WS_HALVES*2 ≈ 617 KB
    auto blocks = [](int n) { return (n + 255) / 256; };

    cvt_f16_kernel<<<blocks(16384),  256, 0, stream>>>(W_s,    w16 + OFF_WS, 16384);
    cvt_wsa_pad_kernel<<<blocks(128 * 160), 256, 0, stream>>>(W_sa, w16 + OFF_WSA);
    cvt_f16_kernel<<<blocks(16384),  256, 0, stream>>>(W_k,    w16 + OFF_WK, 16384);
    cvt_f16_kernel<<<blocks(16384),  256, 0, stream>>>(W_q,    w16 + OFF_WQ, 16384);
    cvt_f16_kernel<<<blocks(16384),  256, 0, stream>>>(W_v,    w16 + OFF_WV, 16384);
    cvt_f16_kernel<<<blocks(16384),  256, 0, stream>>>(W_e,    w16 + OFF_WE, 16384);
    cvt_f16_kernel<<<blocks(196608), 256, 0, stream>>>(gru_Wi, w16 + OFF_WI, 196608);
    cvt_f16_kernel<<<blocks(16384),  256, 0, stream>>>(W1,     w16 + OFF_W1, 16384);

    critic_fused_kernel<<<1024, 256, 0, stream>>>(
        states, actions, w16,
        b_sa, b_s, b_k, b_q, b_v, b_e,
        gru_bi, gru_bh, b1, W2, b2,
        (float*)d_out);
}
